// MixedConvWithReLU_28398323761155
// MI455X (gfx1250) — matrix operand order repaired and checked by reading, not compiled, not run on hardware
//
#include <hip/hip_runtime.h>
#include <hip/hip_bf16.h>

typedef _Float16 half16 __attribute__((ext_vector_type(16)));
typedef float    float8 __attribute__((ext_vector_type(8)));

// Problem constants (from reference setup_inputs)
#define BATCH   32
#define CIN     256
#define COUT    256
#define HH      56
#define WW      56
#define HW      (HH*WW)         // 3136
#define KDIM    2304            // CIN * 9
#define KTILES  72              // 2304 / 32
#define NTOT    (BATCH*HW)      // 100352
#define NBLKS   (NTOT/32)       // 3136
#define PER_BR  (COUT*KDIM)     // 589824 weights per branch
#define FRAG_HALFS   512                    // one 16m x 32k A fragment
#define MT_STRIDE    (KTILES*FRAG_HALFS)    // 36864 halfs per m-tile
#define PACKA_HALFS  (3*16*MT_STRIDE)
#define PACKA_BYTES  (PACKA_HALFS*2)        // 3,538,944 bytes

// ---------------------------------------------------------------------------
// 1) zero the scale accumulators in workspace
__global__ __launch_bounds__(64) void init_scal_kernel(float* scal) {
  if (threadIdx.x < 3) scal[threadIdx.x] = 0.0f;   // 0.0f bits == 0u for max too
}

// ---------------------------------------------------------------------------
// 2) per-branch reduction: branch0 -> sum(|W|), branch1/2 -> max(|W|)
__global__ __launch_bounds__(256) void reduce_kernel(const float* __restrict__ W,
                                                     float* __restrict__ scal) {
  __shared__ float red[256];
  const int br  = blockIdx.x / 2304;               // 2304 blocks per branch
  const int idx = br * PER_BR + (blockIdx.x % 2304) * 256 + threadIdx.x;
  red[threadIdx.x] = fabsf(W[idx]);
  __syncthreads();
  if (br == 0) {
    for (int s = 128; s > 0; s >>= 1) {
      if (threadIdx.x < s) red[threadIdx.x] += red[threadIdx.x + s];
      __syncthreads();
    }
    if (threadIdx.x == 0) atomicAdd(&scal[0], red[0]);
  } else {
    for (int s = 128; s > 0; s >>= 1) {
      if (threadIdx.x < s) red[threadIdx.x] = fmaxf(red[threadIdx.x], red[threadIdx.x + s]);
      __syncthreads();
    }
    if (threadIdx.x == 0)
      atomicMax((unsigned int*)&scal[br], __float_as_uint(red[0]));
  }
}

// ---------------------------------------------------------------------------
// 3) quantize weights and pack into WMMA 16-bit A-fragment layout:
//    frag slot = [br][mt(16)][kt(72)][lane(32)][j(16)] halfs.
//    (lane,j) -> (m,k): m = lane&15; lanes 0-15 hold k in {0-7,16-23},
//    lanes 16-31 hold k in {8-15,24-31}; j<8 -> low K half, j>=8 -> +16.
__global__ __launch_bounds__(256) void pack_kernel(const float* __restrict__ W,
                                                   const float* __restrict__ scal,
                                                   _Float16* __restrict__ Apk) {
  const int idx = blockIdx.x * 256 + threadIdx.x;  // < 3*589824
  const int br  = idx / PER_BR;
  const int rem = idx - br * PER_BR;
  const int m   = rem / KDIM;
  const int k   = rem - m * KDIM;
  const float w = W[idx];
  float q;
  if (br == 0) {                                   // 1-bit: sign(w)*mean|w|
    const float s = scal[0] * (1.0f / (float)PER_BR);
    q = (w > 0.f) ? s : ((w < 0.f) ? -s : 0.f);
  } else {                                         // symmetric uniform
    const float mx   = scal[br];
    const float lev  = (br == 1) ? 7.0f : 32767.0f;
    const float step = mx / lev;
    q = (step > 0.f) ? rintf(w / step) * step : 0.f;
  }
  const int mt = m >> 4, ml = m & 15;
  const int kt = k >> 5, kin = k & 31;
  const int hik  = kin >> 4;                       // K>=16 inside tile
  const int rem8 = kin & 15;
  const int lane = ml + ((rem8 >= 8) ? 16 : 0);
  const int j    = (hik ? 8 : 0) + (rem8 & 7);
  Apk[(((br * 16 + mt) * KTILES + kt) * 32 + lane) * 16 + j] = (_Float16)q;
}

// ---------------------------------------------------------------------------
// incremental im2col gather state: one thread supplies 4 f16 elements per step
struct GatherState {
  int ciS[4];      // ci*3136 (channel offset into x image plane)
  int r9[4];       // k mod 9
  int base;        // bb*CIN*HW + (oh-1)*WW + (ow-1)
  int oh, ow;
  int lds_off[4];  // invariant LDS half-offsets inside one 1024-half buffer
};

__device__ __forceinline__ void gather_step(const float* __restrict__ x,
                                            _Float16* __restrict__ sbuf,
                                            GatherState& st) {
#pragma unroll
  for (int e = 0; e < 4; ++e) {
    int r9 = st.r9[e];
    const int kh = (r9 >= 6) ? 2 : ((r9 >= 3) ? 1 : 0);
    const int kw = r9 - kh * 3;
    const int ih = st.oh + kh - 1;
    const int iw = st.ow + kw - 1;
    float v = 0.0f;
    if ((unsigned)ih < (unsigned)HH && (unsigned)iw < (unsigned)WW)
      v = x[st.base + st.ciS[e] + kh * WW + kw];
    sbuf[st.lds_off[e]] = (_Float16)v;
    // advance k by 32: ci += 3, (k mod 9) += 5 with carry
    r9 += 5;
    int c = st.ciS[e] + 3 * HW;
    if (r9 >= 9) { r9 -= 9; c += HW; }
    st.r9[e] = r9;
    st.ciS[e] = c;
  }
}

// ---------------------------------------------------------------------------
// 4) fused implicit-GEMM: per block M=256 x N=32, K-loop over 72 steps of 32.
//    LDS double-buffered B (one barrier/step); gather for kt+1 overlaps the
//    12 WMMAs of kt. Epilogue: BN + act_quant + softmax blend.
__global__ __launch_bounds__(256) void fused_gemm_kernel(
    const float* __restrict__ x,
    const _Float16* __restrict__ Apk,
    const float* __restrict__ bn_gamma, const float* __restrict__ bn_beta,
    const float* __restrict__ bn_mean,  const float* __restrict__ bn_var,
    const float* __restrict__ alphas,
    float* __restrict__ out) {

  __shared__ __align__(16) _Float16 sB[2][1024];   // double-buffered B tiles
  __shared__ float sInv[768];
  __shared__ float sBias[768];

  const int tid  = threadIdx.x;
  const int lane = tid & 31;
  const int wave = tid >> 5;
  const int n_base = blockIdx.x * 32;

  // stage fused BN parameters (3 branches x 256 channels)
  for (int i = tid; i < 768; i += 256) {
    const float inv = bn_gamma[i] * rsqrtf(bn_var[i] + 1e-5f);
    sInv[i]  = inv;
    sBias[i] = bn_beta[i] - bn_mean[i] * inv;
  }

  float8 acc[2][2][3];
#pragma unroll
  for (int a = 0; a < 2; ++a)
#pragma unroll
    for (int b = 0; b < 2; ++b)
#pragma unroll
      for (int c = 0; c < 3; ++c) acc[a][b][c] = (float8){};

  // ---- init incremental gather state (n fixed per thread) ----
  GatherState st;
  {
    const int n_local = tid & 31;
    const int krow    = tid >> 5;                  // 0..7
    const int n  = n_base + n_local;
    const int bb = n / HW;
    const int rN = n - bb * HW;
    st.oh = rN / WW;
    st.ow = rN - st.oh * WW;
    st.base = bb * CIN * HW + (st.oh - 1) * WW + (st.ow - 1);
    const int nt_g    = n_local >> 4;
    const int lane_lo = n_local & 15;
#pragma unroll
    for (int e = 0; e < 4; ++e) {
      const int k0 = e * 8 + krow;                 // k_local 0..31
      const int ci = k0 / 9;
      st.r9[e]  = k0 - ci * 9;
      st.ciS[e] = ci * HW;
      st.lds_off[e] = nt_g * 512 + (((k0 & 8) << 1) + lane_lo) * 16 + (k0 & 7) + ((k0 & 16) >> 1);
    }
  }

  // A fragments: single base pointer + compile-time immediate offsets
  const _Float16* aptr = Apk + (wave * 2) * MT_STRIDE + lane * 16;

  // prologue: fill buffer 0 for kt=0
  gather_step(x, sB[0], st);
  __syncthreads();

  for (int kt = 0; kt < KTILES; ++kt) {
    const int cur = kt & 1;

    // B fragments from LDS
    const half16 bf0 = *(const half16*)&sB[cur][lane * 16];
    const half16 bf1 = *(const half16*)&sB[cur][512 + lane * 16];

    // 6 A-fragment loads as one clause from single base + immediates
    half16 af[2][3];
#pragma unroll
    for (int mi = 0; mi < 2; ++mi)
#pragma unroll
      for (int br = 0; br < 3; ++br)
        af[mi][br] = *(const half16*)&aptr[(br * 16 + mi) * MT_STRIDE];
    aptr += FRAG_HALFS;

    // gather next K-step into other buffer; overlaps with WMMAs below
    if (kt + 1 < KTILES) gather_step(x, sB[cur ^ 1], st);

#pragma unroll
    for (int mi = 0; mi < 2; ++mi)
#pragma unroll
      for (int br = 0; br < 3; ++br) {
        acc[mi][0][br] = __builtin_amdgcn_wmma_f32_16x16x32_f16(
            false, af[mi][br], false, bf0, (short)0, acc[mi][0][br], false, false);
        acc[mi][1][br] = __builtin_amdgcn_wmma_f32_16x16x32_f16(
            false, af[mi][br], false, bf1, (short)0, acc[mi][1][br], false, false);
      }

    __syncthreads();   // gather stores visible before next step's frag loads
  }

  // ---- epilogue: softmax(alphas) blend of BN + act_quant branches ----
  const float a0 = alphas[0], a1 = alphas[1], a2 = alphas[2];
  const float am = fmaxf(a0, fmaxf(a1, a2));
  const float e0 = __expf(a0 - am), e1 = __expf(a1 - am), e2 = __expf(a2 - am);
  const float ei = 1.0f / (e0 + e1 + e2);
  const float wsm[3]  = {e0 * ei, e1 * ei, e2 * ei};
  const float lev[3]  = {1.0f, 15.0f, 65535.0f};
  const float rlev[3] = {1.0f, 1.0f / 15.0f, 1.0f / 65535.0f};

  const int hi = (lane >> 4) << 3;                 // +8 row offset for lanes 16-31
#pragma unroll
  for (int nt = 0; nt < 2; ++nt) {
    const int n_o  = n_base + nt * 16 + (lane & 15);
    const int b_o  = n_o / HW;
    const int r_o  = n_o - b_o * HW;
    const int oh_o = r_o / WW;
    const int ow_o = r_o - oh_o * WW;
#pragma unroll
    for (int mi = 0; mi < 2; ++mi) {
      const int mt = wave * 2 + mi;
#pragma unroll
      for (int r = 0; r < 8; ++r) {
        const int c = mt * 16 + r + hi;
        float o = 0.0f;
#pragma unroll
        for (int br = 0; br < 3; ++br) {
          float y = acc[mi][nt][br][r] * sInv[br * 256 + c] + sBias[br * 256 + c];
          y = fminf(fmaxf(y, 0.0f), 1.0f);         // clamp [0,1]
          y = rintf(y * lev[br]) * rlev[br];       // round-half-even like jnp.round
          o += wsm[br] * y;
        }
        out[((b_o * COUT + c) * HH + oh_o) * WW + ow_o] = o;
      }
    }
  }
}

// ---------------------------------------------------------------------------
extern "C" void kernel_launch(void* const* d_in, const int* in_sizes, int n_in,
                              void* d_out, int out_size, void* d_ws, size_t ws_size,
                              hipStream_t stream) {
  (void)in_sizes; (void)n_in; (void)out_size; (void)ws_size;
  const float* x        = (const float*)d_in[0];
  const float* W        = (const float*)d_in[1];
  const float* bn_gamma = (const float*)d_in[2];
  const float* bn_beta  = (const float*)d_in[3];
  const float* bn_mean  = (const float*)d_in[4];
  const float* bn_var   = (const float*)d_in[5];
  const float* alphas   = (const float*)d_in[6];
  float* out = (float*)d_out;

  _Float16* Apk = (_Float16*)d_ws;
  float*    scal = (float*)((char*)d_ws + PACKA_BYTES);

  init_scal_kernel<<<1, 64, 0, stream>>>(scal);
  reduce_kernel<<<3 * 2304, 256, 0, stream>>>(W, scal);
  pack_kernel<<<3 * 2304, 256, 0, stream>>>(W, scal, Apk);
  fused_gemm_kernel<<<NBLKS, 256, 0, stream>>>(x, Apk, bn_gamma, bn_beta,
                                               bn_mean, bn_var, alphas, out);
}
